// T5Attention_4922032521748
// MI455X (gfx1250) — matrix operand order repaired and checked by reading, not compiled, not run on hardware
//
#include <hip/hip_runtime.h>
#include <hip/hip_bf16.h>

// ---------------------------------------------------------------------------
// T5 encoder self-attention, MI455X (gfx1250, wave32, WMMA f16 16x16x32).
// Pipeline: cast -> weight-transpose -> bias LUT -> QKV wmma-GEMM (32x64
// wave tiles) -> flash attention (64-key slabs: 8 wmma QK^T + online softmax
// + 8 wmma P@V) -> output wmma-GEMM (f32 out).
// ---------------------------------------------------------------------------

typedef __attribute__((ext_vector_type(16))) _Float16 v16h;
typedef __attribute__((ext_vector_type(8)))  _Float16 v8h;
typedef __attribute__((ext_vector_type(8)))  float    v8f;

#define WMMA16(a, b, c) \
  __builtin_amdgcn_wmma_f32_16x16x32_f16(false, (a), false, (b), (short)0, (c), false, false)

namespace {
constexpr int kB  = 2;
constexpr int kS  = 2048;
constexpr int kD  = 1024;
constexpr int kH  = 16;
constexpr int kDK = 64;
constexpr int kHD = kH * kDK;        // 1024
constexpr int kRows = kB * kS;       // 4096
constexpr int kREL  = 2 * kS - 1;    // 4095 relative distances
}

// A-fragment (16x32 f16): lane m = l%16; elems 0..7 = K[hi*8 .. hi*8+7],
// elems 8..15 = K[16+hi*8 .. 16+hi*8+7]  (ISA 7.12.2 16-bit A table)
__device__ __forceinline__ v16h load_a_frag(const _Float16* __restrict__ rowPtr, int hi) {
  v8h r0 = *(const v8h*)(rowPtr + hi * 8);
  v8h r1 = *(const v8h*)(rowPtr + 16 + hi * 8);
  v16h a;
#pragma unroll
  for (int i = 0; i < 8; ++i) { a[i] = r0[i]; a[8 + i] = r1[i]; }
  return a;
}

// B-fragment (32x16 f16): lane n = l%16; elems 0..7 = K[hi*8 .. hi*8+7],
// elems 8..15 = K[16+hi*8 .. 16+hi*8+7] (same k order as A fragment).
__device__ __forceinline__ v16h load_b_frag(const _Float16* __restrict__ colPtr, int hi) {
  v8h r0 = *(const v8h*)(colPtr + hi * 8);
  v8h r1 = *(const v8h*)(colPtr + 16 + hi * 8);
  v16h b;
#pragma unroll
  for (int i = 0; i < 8; ++i) { b[i] = r0[i]; b[8 + i] = r1[i]; }
  return b;
}

// --------------------------- small prep kernels ----------------------------

__global__ void cast_x_kernel(const float* __restrict__ X, _Float16* __restrict__ Xh, int n) {
  int t = blockIdx.x * blockDim.x + threadIdx.x;
  if (t < n) Xh[t] = (_Float16)X[t];
}

__global__ void wtrans_kernel(const float* __restrict__ W0, const float* __restrict__ W1,
                              const float* __restrict__ W2, const float* __restrict__ W3,
                              _Float16* __restrict__ T0, _Float16* __restrict__ T1,
                              _Float16* __restrict__ T2, _Float16* __restrict__ T3) {
  int n = blockIdx.x * 16 + threadIdx.x;   // output row of W^T
  int k = blockIdx.y * 16 + threadIdx.y;   // output col of W^T
  const float* W = (blockIdx.z == 0) ? W0 : (blockIdx.z == 1) ? W1 : (blockIdx.z == 2) ? W2 : W3;
  _Float16*    T = (blockIdx.z == 0) ? T0 : (blockIdx.z == 1) ? T1 : (blockIdx.z == 2) ? T2 : T3;
  if (n < kD && k < kD) T[(size_t)n * kD + k] = (_Float16)W[(size_t)k * kD + n];
}

// bias LUT: tab[h*kREL + (rel + S - 1)] = rel_bias[bucket(rel)][h]
__global__ void bias_table_kernel(const float* __restrict__ rel_bias, float* __restrict__ tab) {
  int t = blockIdx.x * blockDim.x + threadIdx.x;
  if (t >= kH * kREL) return;
  int h   = t / kREL;
  int rel = (t % kREL) - (kS - 1);
  int bucket = (rel > 0) ? 16 : 0;            // NUM_BUCKETS/2 = 16
  int rp = rel < 0 ? -rel : rel;
  if (rp < 8) {                               // max_exact = 8
    bucket += rp;
  } else {
    int lb = 8 + (int)(logf((float)rp * 0.125f) / logf(16.0f) * 8.0f);
    bucket += (lb < 15) ? lb : 15;
  }
  tab[t] = rel_bias[bucket * kH + h];
}

// --------------------------- QKV projection GEMM ---------------------------
// C[32x64] per wave (2 A-frags x 4 B-frags -> 8 WMMAs / 32-k step).
// A = Xh [4096x1024] f16, Bt = W^T [1024x1024] f16.
// Epilogue: Q,K -> [B,H,S,DK] f16 ; V -> transposed [B,H,DK,S] f16.
__global__ void __launch_bounds__(256)
gemm_qkv_kernel(const _Float16* __restrict__ Xh,
                const _Float16* __restrict__ WtQ, const _Float16* __restrict__ WtK,
                const _Float16* __restrict__ WtV,
                _Float16* __restrict__ Qb, _Float16* __restrict__ Kb,
                _Float16* __restrict__ VTb) {
  const int lane = threadIdx.x & 31;
  const int wave = threadIdx.x >> 5;
  const int hi   = lane >> 4;
  const int ln   = lane & 15;
  const int m0   = (blockIdx.y * 8 + wave) * 32;
  const int n0   = blockIdx.x * 64;
  const int mat  = blockIdx.z;
  const _Float16* __restrict__ Wt = (mat == 0) ? WtQ : (mat == 1) ? WtK : WtV;

  v8f acc[2][4] = {};
  const _Float16* aRow0 = Xh + (size_t)(m0 + ln) * kD;
  const _Float16* aRow1 = Xh + (size_t)(m0 + 16 + ln) * kD;
  for (int k = 0; k < kD; k += 32) {
    v16h a0 = load_a_frag(aRow0 + k, hi);
    v16h a1 = load_a_frag(aRow1 + k, hi);
#pragma unroll
    for (int j = 0; j < 4; ++j) {
      const _Float16* bCol = Wt + (size_t)(n0 + j * 16 + ln) * kD + k;
      v16h b = load_b_frag(bCol, hi);
      acc[0][j] = WMMA16(a0, b, acc[0][j]);
      acc[1][j] = WMMA16(a1, b, acc[1][j]);
    }
  }

  const int h = n0 >> 6;  // entire wave tile lies in one head
#pragma unroll
  for (int p = 0; p < 2; ++p) {
#pragma unroll
    for (int j = 0; j < 4; ++j) {
      const int d = j * 16 + ln;
#pragma unroll
      for (int i = 0; i < 8; ++i) {
        int r = m0 + p * 16 + i + hi * 8;   // global row = b*S + s
        int b = r >> 11, s = r & (kS - 1);
        _Float16 val = (_Float16)acc[p][j][i];
        size_t bh = (size_t)(b * kH + h);
        if (mat == 0)      Qb [(bh * kS + s) * kDK + d] = val;
        else if (mat == 1) Kb [(bh * kS + s) * kDK + d] = val;
        else               VTb[(bh * kDK + d) * kS + s] = val;
      }
    }
  }
}

// ------------------------------- attention ---------------------------------
// One wave = one 16-query tile of one (b,h). Streams 64 keys/iter:
// 8 WMMAs for QK^T, online softmax in f32 (shuffle reductions amortized over
// 4 score tiles), P through LDS (C->A layout), 8 WMMAs for P@V with
// accumulator rescale. Output -> AO[B*S, H*DK] f16.
__global__ void __launch_bounds__(256)
attn_kernel(const _Float16* __restrict__ Qb, const _Float16* __restrict__ Kb,
            const _Float16* __restrict__ VTb, const float* __restrict__ biasTab,
            _Float16* __restrict__ AO) {
  __shared__ alignas(32) _Float16 sP[8][16][64];

  const int lane = threadIdx.x & 31;
  const int wave = threadIdx.x >> 5;
  const int hi   = lane >> 4;
  const int ln   = lane & 15;
  const int b    = blockIdx.z;
  const int h    = blockIdx.y;
  const int q0   = (blockIdx.x * 8 + wave) * 16;

  const size_t bh = (size_t)(b * kH + h);
  const _Float16* __restrict__ Qp = Qb + bh * kS * kDK;
  const _Float16* __restrict__ Kp = Kb + bh * kS * kDK;
  const _Float16* __restrict__ Vt = VTb + bh * kDK * kS;
  const float* __restrict__ bt = biasTab + (size_t)h * kREL + (kS - 1);

  // Q tile: 16x64 -> two A fragments (d 0..31, d 32..63)
  v16h qa0 = load_a_frag(Qp + (size_t)(q0 + ln) * kDK, hi);
  v16h qa1 = load_a_frag(Qp + (size_t)(q0 + ln) * kDK + 32, hi);

  float m[8], l[8];
#pragma unroll
  for (int i = 0; i < 8; ++i) { m[i] = -1.0e30f; l[i] = 0.0f; }
  v8f acc[4] = {};

  for (int kb = 0; kb < kS; kb += 64) {
    if (kb + 64 < kS) {
      __builtin_prefetch(Kp + (size_t)(kb + 64 + ln) * kDK, 0, 1);
      __builtin_prefetch(Vt + (size_t)ln * kS + kb + 64, 0, 1);
    }

    // scores: S = Q K^T  (four 16x16 tiles, contraction over DK=64)
    v8f st[4];
#pragma unroll
    for (int t = 0; t < 4; ++t) {
      const int key = kb + t * 16 + ln;
      v16h ka = load_b_frag(Kp + (size_t)key * kDK, hi);
      v16h kc = load_b_frag(Kp + (size_t)key * kDK + 32, hi);
      v8f s = {};
      s = WMMA16(qa0, ka, s);
      s = WMMA16(qa1, kc, s);
      st[t] = s;
    }

    float alpha[8];
#pragma unroll
    for (int i = 0; i < 8; ++i) {
      const int q = q0 + i + hi * 8;               // C layout: row = i + 8*hiHalf
#pragma unroll
      for (int t = 0; t < 4; ++t) st[t][i] += bt[(kb + t * 16 + ln) - q];
      // row max across the 16 lanes of this half-wave (cols of the tiles)
      float v = fmaxf(fmaxf(st[0][i], st[1][i]), fmaxf(st[2][i], st[3][i]));
      v = fmaxf(v, __shfl_xor(v, 8, 32));
      v = fmaxf(v, __shfl_xor(v, 4, 32));
      v = fmaxf(v, __shfl_xor(v, 2, 32));
      v = fmaxf(v, __shfl_xor(v, 1, 32));
      float mn = fmaxf(m[i], v);
      alpha[i] = __expf(m[i] - mn);
      m[i] = mn;
      float rs = 0.0f;
#pragma unroll
      for (int t = 0; t < 4; ++t) {
        float p = __expf(st[t][i] - mn);
        st[t][i] = p;
        rs += p;
        sP[wave][i + hi * 8][t * 16 + ln] = (_Float16)p;  // [row][key-in-slab]
      }
      rs += __shfl_xor(rs, 8, 32);
      rs += __shfl_xor(rs, 4, 32);
      rs += __shfl_xor(rs, 2, 32);
      rs += __shfl_xor(rs, 1, 32);
      l[i] = l[i] * alpha[i] + rs;
    }
    __syncthreads();  // uniform trip count across all waves

    // reload P as two A fragments (16x32 each: keys 0..31 and 32..63 of slab)
    v16h pf0, pf1;
    {
      const _Float16* pr = &sP[wave][ln][0];
      v8h r0 = *(const v8h*)(pr + hi * 8);
      v8h r1 = *(const v8h*)(pr + 16 + hi * 8);
      v8h r2 = *(const v8h*)(pr + 32 + hi * 8);
      v8h r3 = *(const v8h*)(pr + 48 + hi * 8);
#pragma unroll
      for (int i = 0; i < 8; ++i) {
        pf0[i] = r0[i]; pf0[8 + i] = r1[i];
        pf1[i] = r2[i]; pf1[8 + i] = r3[i];
      }
    }

    // O = O*alpha + P @ V   (V^T layout -> contiguous B fragments)
#pragma unroll
    for (int j = 0; j < 4; ++j) {
#pragma unroll
      for (int i = 0; i < 8; ++i) acc[j][i] *= alpha[i];
      v16h vb0 = load_b_frag(Vt + (size_t)(j * 16 + ln) * kS + kb, hi);
      v16h vb1 = load_b_frag(Vt + (size_t)(j * 16 + ln) * kS + kb + 32, hi);
      acc[j] = WMMA16(pf0, vb0, acc[j]);
      acc[j] = WMMA16(pf1, vb1, acc[j]);
    }
  }

  // normalize and write AO[B*S, H*DK] as f16
#pragma unroll
  for (int j = 0; j < 4; ++j) {
#pragma unroll
    for (int i = 0; i < 8; ++i) {
      const int q = q0 + i + hi * 8;
      float o = acc[j][i] / l[i];
      AO[((size_t)(b * kS + q)) * kHD + h * kDK + j * 16 + ln] = (_Float16)o;
    }
  }
}

// ------------------------------ output GEMM --------------------------------
// C[32x64] per wave, f32 out.
__global__ void __launch_bounds__(256)
gemm_out_kernel(const _Float16* __restrict__ AO, const _Float16* __restrict__ WtO,
                float* __restrict__ out) {
  const int lane = threadIdx.x & 31;
  const int wave = threadIdx.x >> 5;
  const int hi   = lane >> 4;
  const int ln   = lane & 15;
  const int m0   = (blockIdx.y * 8 + wave) * 32;
  const int n0   = blockIdx.x * 64;

  v8f acc[2][4] = {};
  const _Float16* aRow0 = AO + (size_t)(m0 + ln) * kHD;
  const _Float16* aRow1 = AO + (size_t)(m0 + 16 + ln) * kHD;
  for (int k = 0; k < kHD; k += 32) {
    v16h a0 = load_a_frag(aRow0 + k, hi);
    v16h a1 = load_a_frag(aRow1 + k, hi);
#pragma unroll
    for (int j = 0; j < 4; ++j) {
      const _Float16* bCol = WtO + (size_t)(n0 + j * 16 + ln) * kHD + k;
      v16h b = load_b_frag(bCol, hi);
      acc[0][j] = WMMA16(a0, b, acc[0][j]);
      acc[1][j] = WMMA16(a1, b, acc[1][j]);
    }
  }
#pragma unroll
  for (int p = 0; p < 2; ++p) {
#pragma unroll
    for (int j = 0; j < 4; ++j) {
#pragma unroll
      for (int i = 0; i < 8; ++i) {
        int r = m0 + p * 16 + i + hi * 8;
        out[(size_t)r * kD + n0 + j * 16 + ln] = acc[p][j][i];
      }
    }
  }
}

// ------------------------------ host launcher ------------------------------
extern "C" void kernel_launch(void* const* d_in, const int* in_sizes, int n_in,
                              void* d_out, int out_size, void* d_ws, size_t ws_size,
                              hipStream_t stream) {
  (void)in_sizes; (void)n_in; (void)out_size; (void)ws_size;
  const float* X        = (const float*)d_in[0];
  const float* Wq       = (const float*)d_in[1];
  const float* Wk       = (const float*)d_in[2];
  const float* Wv       = (const float*)d_in[3];
  const float* Wo       = (const float*)d_in[4];
  const float* rel_bias = (const float*)d_in[5];
  float* out            = (float*)d_out;

  // workspace carve-up (f16 intermediates; ~50.6 MB total)
  char* w = (char*)d_ws;
  const size_t szX  = (size_t)kRows * kD * sizeof(_Float16);         // 8 MB
  const size_t szW  = (size_t)kD * kD * sizeof(_Float16);            // 2 MB each
  const size_t szQ  = (size_t)kB * kH * kS * kDK * sizeof(_Float16); // 8 MB each
  _Float16* Xh  = (_Float16*)w;              w += szX;
  _Float16* WtQ = (_Float16*)w;              w += szW;
  _Float16* WtK = (_Float16*)w;              w += szW;
  _Float16* WtV = (_Float16*)w;              w += szW;
  _Float16* WtO = (_Float16*)w;              w += szW;
  _Float16* Qb  = (_Float16*)w;              w += szQ;
  _Float16* Kb  = (_Float16*)w;              w += szQ;
  _Float16* VTb = (_Float16*)w;              w += szQ;
  _Float16* AO  = (_Float16*)w;              w += szX;
  float*    biasTab = (float*)w;             w += (size_t)kH * kREL * sizeof(float);

  // 1) casts + weight transposes + bias LUT
  {
    int n = kRows * kD;
    cast_x_kernel<<<(n + 255) / 256, 256, 0, stream>>>(X, Xh, n);
    dim3 g(kD / 16, kD / 16, 4), blk(16, 16);
    wtrans_kernel<<<g, blk, 0, stream>>>(Wq, Wk, Wv, Wo, WtQ, WtK, WtV, WtO);
    int nb = kH * kREL;
    bias_table_kernel<<<(nb + 255) / 256, 256, 0, stream>>>(rel_bias, biasTab);
  }

  // 2) Q/K/V projections (wave tile 32x64, 8 waves/block)
  {
    dim3 g(kD / 64, kRows / (32 * 8), 3);
    gemm_qkv_kernel<<<g, 256, 0, stream>>>(Xh, WtQ, WtK, WtV, Qb, Kb, VTb);
  }

  // 3) attention (one wave per 16-query tile, 64-key slabs)
  {
    dim3 g(kS / (16 * 8), kH, kB);
    attn_kernel<<<g, 256, 0, stream>>>(Qb, Kb, VTb, biasTab, AO);
  }

  // 4) output projection -> f32 d_out
  {
    dim3 g(kD / 64, kRows / (32 * 8));
    gemm_out_kernel<<<g, 256, 0, stream>>>(AO, WtO, out);
  }
}
